// UpBlock_6279242186809
// MI455X (gfx1250) — matrix operand order repaired and checked by reading, not compiled, not run on hardware
//
#include <hip/hip_runtime.h>
#include <hip/hip_bf16.h>

// ---------------------------------------------------------------------------
// CDNA5 / gfx1250 sparse-conv UpBlock using V_WMMA_F32_16X16X32_BF16.
// wave32; each wave owns a 32-row x 64-col output tile (8 f32 16x16 accs),
// so each B (weight) fragment load feeds two WMMAs.
// ---------------------------------------------------------------------------

typedef __bf16 bf16_t;
typedef __bf16 v8bf  __attribute__((ext_vector_type(8)));
typedef __bf16 v16bf __attribute__((ext_vector_type(16)));
typedef float  v8f   __attribute__((ext_vector_type(8)));

__device__ __forceinline__ v16bf load_frag(const bf16_t* p0, const bf16_t* p1) {
    v8bf lo = *(const v8bf*)p0;   // 16B aligned
    v8bf hi = *(const v8bf*)p1;   // 16B aligned
    return __builtin_shufflevector(lo, hi, 0,1,2,3,4,5,6,7,8,9,10,11,12,13,14,15);
}

// mode 0: z = lrelu(conv); write bf16 z; accumulate per-channel sum/sumsq.
// mode 1: z = conv + skip; write bf16 z; no stats.
__global__ __launch_bounds__(128) void subm_conv_wmma_kernel(
    const bf16_t* __restrict__ feats, int nIn, int Cin,
    const int*    __restrict__ nbr, int Kofs, int Nout,
    const bf16_t* __restrict__ Wt,          // [Kofs][64][Cin]  (Cout-major, K-contig)
    const float*  __restrict__ skip,        // mode 1 only
    bf16_t*       __restrict__ zout,        // [Nout][64]
    float*        __restrict__ stats,       // [128] = {sum[64], sumsq[64]}  (mode 0)
    int mode)
{
    __shared__ float s_sum[64];
    __shared__ float s_sq[64];
    const int tid = threadIdx.x;
    if (tid < 64) { s_sum[tid] = 0.0f; s_sq[tid] = 0.0f; }
    __syncthreads();

    const int lane = tid & 31;
    const int wave = tid >> 5;
    const int half = lane >> 4;      // K-group select (A) / K-group select (B)
    const int mrow = lane & 15;      // A row / B column within tile
    const int rowBase = (blockIdx.x * 4 + wave) * 32;   // two 16-row M-tiles
    const int r0 = rowBase + mrow;
    const int r1 = rowBase + 16 + mrow;

    v8f acc[8];
    #pragma unroll
    for (int i = 0; i < 8; ++i) acc[i] = (v8f){};

    for (int k = 0; k < Kofs; ++k) {
        const int* nrow = nbr + (size_t)k * (size_t)Nout;
        int idx0 = nIn, idx1 = nIn;                     // sentinel -> zero row
        if (r0 < Nout) idx0 = nrow[r0];
        if (r1 < Nout) idx1 = nrow[r1];
        const bool live0 = ((unsigned)idx0 < (unsigned)nIn);
        const bool live1 = ((unsigned)idx1 < (unsigned)nIn);
        const bf16_t* arow0 = feats + (size_t)idx0 * (size_t)Cin;
        const bf16_t* arow1 = feats + (size_t)idx1 * (size_t)Cin;
        const bf16_t* wk    = Wt + (size_t)k * 64u * (size_t)Cin;

        for (int cc = 0; cc < Cin; cc += 32) {
            // --- A fragments: 16x32 bf16, row = mrow, K-groups by lane half ---
            v16bf a0 = {}, a1 = {};
            if (live0) a0 = load_frag(arow0 + cc + half * 8, arow0 + cc + 16 + half * 8);
            if (live1) a1 = load_frag(arow1 + cc + half * 8, arow1 + cc + 16 + half * 8);

            // --- 4 B fragments (32x16 each) covering Cout = 64; each feeds 2 WMMAs ---
            const bf16_t* wc = wk + (size_t)mrow * (size_t)Cin + cc + half * 8;
            #pragma unroll
            for (int t = 0; t < 4; ++t) {
                v16bf b = load_frag(wc, wc + 16);
                acc[t]     = __builtin_amdgcn_wmma_f32_16x16x32_bf16(
                                 false, a0, false, b, (short)0, acc[t],     false, false);
                acc[4 + t] = __builtin_amdgcn_wmma_f32_16x16x32_bf16(
                                 false, a1, false, b, (short)0, acc[4 + t], false, false);
                wc += 16 * (size_t)Cin;
            }
        }
    }

    // --- epilogue: C/D layout lane0-15 -> M=e, lane16-31 -> M=8+e, col=lane&15 ---
    #pragma unroll
    for (int m = 0; m < 2; ++m) {
        #pragma unroll
        for (int t = 0; t < 4; ++t) {
            const int col = t * 16 + mrow;
            float ls = 0.0f, ls2 = 0.0f;
            #pragma unroll
            for (int e = 0; e < 8; ++e) {
                const int row = rowBase + m * 16 + half * 8 + e;
                if (row < Nout) {
                    float v = acc[m * 4 + t][e];
                    if (mode == 0) {
                        v = (v >= 0.0f) ? v : 0.01f * v;   // LeakyReLU
                        ls += v; ls2 += v * v;
                    } else {
                        v += skip[(size_t)row * 64u + col];
                    }
                    zout[(size_t)row * 64u + col] = (bf16_t)v;
                }
            }
            if (mode == 0) {
                atomicAdd(&s_sum[col], ls);   // ds_add_f32
                atomicAdd(&s_sq[col],  ls2);
            }
        }
    }
    __syncthreads();
    if (mode == 0 && tid < 64) {
        atomicAdd(&stats[tid],      s_sum[tid]);
        atomicAdd(&stats[64 + tid], s_sq[tid]);
    }
}

// ---------------------------------------------------------------------------
// Utility kernels
// ---------------------------------------------------------------------------
__global__ void zero_f32_kernel(float* p, int n) {
    int i = blockIdx.x * blockDim.x + threadIdx.x;
    if (i < n) p[i] = 0.0f;
}

__global__ void cvt_bf16_kernel(const float* __restrict__ src, bf16_t* __restrict__ dst, size_t n) {
    size_t i = (size_t)blockIdx.x * blockDim.x + threadIdx.x;
    size_t stride = (size_t)gridDim.x * blockDim.x;
    for (; i < n; i += stride) dst[i] = (bf16_t)src[i];
}

// W[k][c][d]  ->  Wt[k][d][c]  (bf16), so K-dim is contiguous per output column
__global__ void wt_transpose_kernel(const float* __restrict__ W, bf16_t* __restrict__ Wt,
                                    int K, int C, int D) {
    size_t n = (size_t)K * C * D;
    size_t i = (size_t)blockIdx.x * blockDim.x + threadIdx.x;
    size_t stride = (size_t)gridDim.x * blockDim.x;
    for (; i < n; i += stride) {
        int d = (int)(i % (size_t)D);
        size_t rr = i / (size_t)D;
        int c = (int)(rr % (size_t)C);
        int k = (int)(rr / (size_t)C);
        Wt[((size_t)k * D + d) * (size_t)C + c] = (bf16_t)W[i];
    }
}

// stats -> per-channel scale/bias (training BN, biased variance)
__global__ void bn_prep_kernel(const float* __restrict__ stats,
                               const float* __restrict__ gamma,
                               const float* __restrict__ beta,
                               float invN, float* __restrict__ sb) {
    int c = threadIdx.x;
    if (c < 64) {
        float m   = stats[c] * invN;
        float var = stats[64 + c] * invN - m * m;
        float sc  = gamma[c] * rsqrtf(var + 1e-5f);
        sb[c]      = sc;
        sb[64 + c] = beta[c] - m * sc;
    }
}

__global__ void bn_apply_kernel(const bf16_t* __restrict__ z, const float* __restrict__ sb,
                                size_t n, bf16_t* __restrict__ obf, float* __restrict__ of32) {
    size_t i = (size_t)blockIdx.x * blockDim.x + threadIdx.x;
    size_t stride = (size_t)gridDim.x * blockDim.x;
    for (; i < n; i += stride) {
        int c = (int)(i & 63u);
        float v = (float)z[i] * sb[c] + sb[64 + c];
        if (obf)  obf[i]  = (bf16_t)v;
        if (of32) of32[i] = v;
    }
}

// ---------------------------------------------------------------------------
// Host-side orchestration
// ---------------------------------------------------------------------------
extern "C" void kernel_launch(void* const* d_in, const int* in_sizes, int n_in,
                              void* d_out, int out_size, void* d_ws, size_t ws_size,
                              hipStream_t stream) {
    (void)n_in; (void)out_size; (void)ws_size;

    const float* x      = (const float*)d_in[0];
    const float* skip   = (const float*)d_in[1];
    const int*   nbr_t  = (const int*)d_in[2];
    const int*   nbr_up = (const int*)d_in[3];
    const int*   nbr1   = (const int*)d_in[4];
    const int*   nbr2   = (const int*)d_in[5];
    const int*   nbr3   = (const int*)d_in[6];
    const float* W_t    = (const float*)d_in[7];
    const float* W_up   = (const float*)d_in[8];
    const float* W1     = (const float*)d_in[9];
    const float* W2     = (const float*)d_in[10];
    const float* W3     = (const float*)d_in[11];
    const float* g_t = (const float*)d_in[12]; const float* b_t = (const float*)d_in[13];
    const float* g1  = (const float*)d_in[14]; const float* b1  = (const float*)d_in[15];
    const float* g2  = (const float*)d_in[16]; const float* b2  = (const float*)d_in[17];
    const float* g3  = (const float*)d_in[18]; const float* b3  = (const float*)d_in[19];

    const int Nin  = in_sizes[0] / 128;   // 80000
    const int Nout = in_sizes[1] / 64;    // 200000
    const int Cin  = 128, Cout = 64;

    // ---- carve workspace (256B aligned) ----
    char* cur = (char*)d_ws;
    auto carve = [&](size_t bytes) {
        void* r = (void*)cur;
        cur += (bytes + 255u) & ~(size_t)255u;
        return r;
    };
    bf16_t* xb  = (bf16_t*)carve((size_t)Nin * Cin * sizeof(bf16_t));
    bf16_t* WtT = (bf16_t*)carve((size_t)27 * Cout * Cin  * sizeof(bf16_t));
    bf16_t* WtU = (bf16_t*)carve((size_t)27 * Cout * Cout * sizeof(bf16_t));
    bf16_t* Wt1 = (bf16_t*)carve((size_t)9  * Cout * Cout * sizeof(bf16_t));
    bf16_t* Wt2 = (bf16_t*)carve((size_t)9  * Cout * Cout * sizeof(bf16_t));
    bf16_t* Wt3 = (bf16_t*)carve((size_t)27 * Cout * Cout * sizeof(bf16_t));
    bf16_t* P0  = (bf16_t*)carve((size_t)Nout * Cout * sizeof(bf16_t));
    bf16_t* P1  = (bf16_t*)carve((size_t)Nout * Cout * sizeof(bf16_t));
    bf16_t* P2  = (bf16_t*)carve((size_t)Nout * Cout * sizeof(bf16_t));
    float*  st  = (float*) carve(4 * 128 * sizeof(float));   // 4 stages x {sum,sumsq}
    float*  sb  = (float*) carve(4 * 128 * sizeof(float));   // 4 stages x {scale,bias}

    // ---- prep: zero stats, bf16 convert, weight transpose ----
    zero_f32_kernel<<<2, 256, 0, stream>>>(st, 512);
    {
        size_t n = (size_t)Nin * Cin;
        cvt_bf16_kernel<<<4096, 256, 0, stream>>>(x, xb, n);
    }
    wt_transpose_kernel<<<256, 256, 0, stream>>>(W_t,  WtT, 27, Cin,  Cout);
    wt_transpose_kernel<<<256, 256, 0, stream>>>(W_up, WtU, 27, Cout, Cout);
    wt_transpose_kernel<<<128, 256, 0, stream>>>(W1,   Wt1, 9,  Cout, Cout);
    wt_transpose_kernel<<<128, 256, 0, stream>>>(W2,   Wt2, 9,  Cout, Cout);
    wt_transpose_kernel<<<256, 256, 0, stream>>>(W3,   Wt3, 27, Cout, Cout);

    const int gIn  = (Nin  + 127) / 128;   // 128 rows per block (4 waves x 32 rows)
    const int gOut = (Nout + 127) / 128;
    const int gApplyIn  = (int)(((size_t)Nin  * 64 + 255) / 256);
    const int gApplyOut = (int)(((size_t)Nout * 64 + 255) / 256);

    // ---- stage 1: trans_dilao (3x3x3, 128->64) + lrelu + bn ----
    subm_conv_wmma_kernel<<<gIn, 128, 0, stream>>>(xb, Nin, Cin, nbr_t, 27, Nin,
                                                   WtT, nullptr, P0, st + 0, 0);
    bn_prep_kernel<<<1, 64, 0, stream>>>(st + 0, g_t, b_t, 1.0f / (float)Nin, sb + 0);
    bn_apply_kernel<<<gApplyIn, 256, 0, stream>>>(P0, sb + 0, (size_t)Nin * 64, P1, nullptr);

    // ---- stage 2: inverse conv (3x3x3, 64->64) + skip add ----
    subm_conv_wmma_kernel<<<gOut, 128, 0, stream>>>(P1, Nin, Cout, nbr_up, 27, Nout,
                                                    WtU, skip, P0, nullptr, 1);

    // ---- stage 3: conv1 (1x3x3) + lrelu + bn ----
    subm_conv_wmma_kernel<<<gOut, 128, 0, stream>>>(P0, Nout, Cout, nbr1, 9, Nout,
                                                    Wt1, nullptr, P2, st + 128, 0);
    bn_prep_kernel<<<1, 64, 0, stream>>>(st + 128, g1, b1, 1.0f / (float)Nout, sb + 128);
    bn_apply_kernel<<<gApplyOut, 256, 0, stream>>>(P2, sb + 128, (size_t)Nout * 64, P1, nullptr);

    // ---- stage 4: conv2 (3x1x3) + lrelu + bn ----
    subm_conv_wmma_kernel<<<gOut, 128, 0, stream>>>(P1, Nout, Cout, nbr2, 9, Nout,
                                                    Wt2, nullptr, P0, st + 256, 0);
    bn_prep_kernel<<<1, 64, 0, stream>>>(st + 256, g2, b2, 1.0f / (float)Nout, sb + 256);
    bn_apply_kernel<<<gApplyOut, 256, 0, stream>>>(P0, sb + 256, (size_t)Nout * 64, P2, nullptr);

    // ---- stage 5: conv3 (3x3x3) + lrelu + bn -> f32 output ----
    subm_conv_wmma_kernel<<<gOut, 128, 0, stream>>>(P2, Nout, Cout, nbr3, 27, Nout,
                                                    Wt3, nullptr, P1, st + 384, 0);
    bn_prep_kernel<<<1, 64, 0, stream>>>(st + 384, g3, b3, 1.0f / (float)Nout, sb + 384);
    bn_apply_kernel<<<gApplyOut, 256, 0, stream>>>(P1, sb + 384, (size_t)Nout * 64,
                                                   nullptr, (float*)d_out);
}
